// PointEncoder_46136538694240
// MI455X (gfx1250) — compile-verified, matrix operand order checked by reading, not run on hardware
//
#include <hip/hip_runtime.h>

typedef __attribute__((ext_vector_type(16))) _Float16 v16h;
typedef __attribute__((ext_vector_type(8)))  _Float16 v8h;
typedef __attribute__((ext_vector_type(8)))  float    v8f;

static inline int cdiv(int a, int b) { return (a + b - 1) / b; }

union PkH { _Float16 h[2]; unsigned int u; };

// ---------------------------------------------------------------------------
// LDS-staged GEMM via CDNA5 WMMA: Y[M,NT] = act(X[M,K] @ W[K,NT] (+ bias))
// Block = 4 waves (128 threads) computing a 64x16 output tile.
// NT is a compile-time constant (16/32/64) so the epilogue stores and B-fill
// addressing fold to immediate offsets. Requires M % 64 == 0 (all call sites
// comply: every M is a multiple of 64).
// K loop in steps of 32 with zero-padded f16 tiles staged through LDS:
//   As[64][32] f16 (row-major), Bs[16][32] f16 (W column, K-contiguous).
// Fill uses clamped-address unconditional loads + packed dword LDS stores.
// Fragment feed per ISA 7.12.2: lane L (rc=L&15, hi=L>>4), element i ->
//   k = 16*(i>>3) + 8*hi + (i&7): two contiguous 8-half chunks -> ds_load_b128.
// ---------------------------------------------------------------------------
template<bool RELU, bool BIAS, int NT>
__global__ __launch_bounds__(128)
void gemm_wmma_kernel(const float* __restrict__ X, const float* __restrict__ W,
                      const float* __restrict__ bias, float* __restrict__ Y,
                      int M, int K) {
  __shared__ __align__(16) _Float16 As[64 * 32];
  __shared__ __align__(16) _Float16 Bs[16 * 32];

  const int tid   = threadIdx.x;
  const int wave  = tid >> 5;
  const int lane  = tid & 31;
  const int rc    = lane & 15;
  const int hi    = lane >> 4;
  const int tileM = blockIdx.x * 64;
  const int tileN = blockIdx.y * 16;

  v8f acc = {};
  for (int k0 = 0; k0 < K; k0 += 32) {
    __syncthreads();
    // ---- cooperative A fill: 64 rows x 32 k; 2 elems/thread/iter, packed ----
#pragma unroll
    for (int e = 0; e < 2048; e += 256) {
      const int idx = e + tid * 2;            // even k
      const int row = idx >> 5, k = idx & 31;
      const int gk = k0 + k;
      const int sk0 = gk < K ? gk : (K - 1);
      const int sk1 = (gk + 1) < K ? (gk + 1) : (K - 1);
      const float v0 = X[(size_t)(tileM + row) * K + sk0];
      const float v1 = X[(size_t)(tileM + row) * K + sk1];
      PkH p;
      p.h[0] = (gk     < K) ? (_Float16)v0 : (_Float16)0.0f;
      p.h[1] = (gk + 1 < K) ? (_Float16)v1 : (_Float16)0.0f;
      *(unsigned int*)(&As[row * 32 + k]) = p.u;
    }
    // ---- cooperative B fill (transposed): 16 cols x 32 k, packed ----
#pragma unroll
    for (int e = 0; e < 512; e += 256) {
      const int idx = e + tid * 2;
      const int n = idx >> 5, k = idx & 31;
      const int gk = k0 + k;
      const int sk0 = gk < K ? gk : (K - 1);
      const int sk1 = (gk + 1) < K ? (gk + 1) : (K - 1);
      const float v0 = W[(size_t)sk0 * NT + tileN + n];
      const float v1 = W[(size_t)sk1 * NT + tileN + n];
      PkH p;
      p.h[0] = (gk     < K) ? (_Float16)v0 : (_Float16)0.0f;
      p.h[1] = (gk + 1 < K) ? (_Float16)v1 : (_Float16)0.0f;
      *(unsigned int*)(&Bs[n * 32 + k]) = p.u;
    }
    __syncthreads();

    // ---- fragment feed from LDS: two 16-byte chunks each for A and B ----
    const _Float16* arow = &As[(wave * 16 + rc) * 32];
    const _Float16* brow = &Bs[rc * 32];
    const v8h a_lo = *(const v8h*)(arow + 8 * hi);
    const v8h a_hi = *(const v8h*)(arow + 16 + 8 * hi);
    const v8h b_lo = *(const v8h*)(brow + 8 * hi);
    const v8h b_hi = *(const v8h*)(brow + 16 + 8 * hi);
    v16h a, b;
#pragma unroll
    for (int i = 0; i < 8; ++i) {
      a[i] = a_lo[i]; a[i + 8] = a_hi[i];
      b[i] = b_lo[i]; b[i + 8] = b_hi[i];
    }
    acc = __builtin_amdgcn_wmma_f32_16x16x32_f16(false, a, false, b,
                                                 (short)0, acc, false, false);
  }

  // ---- epilogue (unguarded; M % 64 == 0): row base + immediate strides ----
  const int n = tileN + rc;
  const float bv = BIAS ? bias[n] : 0.0f;
  float* yrow = Y + (size_t)(tileM + wave * 16 + (hi << 3)) * NT + n;
#pragma unroll
  for (int r = 0; r < 8; ++r) {
    float v = acc[r] + bv;
    if (RELU) v = fmaxf(v, 0.0f);
    yrow[r * NT] = v;
  }
}

// ---------------------------------------------------------------------------
// Brute-force KNN: one thread per query point, LDS-tiled source scan,
// per-thread insertion sort of the k best (k = 3 / 8 / 16).
// ---------------------------------------------------------------------------
template<int KN>
__global__ __launch_bounds__(256)
void knn_kernel(const float* __restrict__ Q, const float* __restrict__ S,
                int* __restrict__ out, int Mq, int Ns) {
  const int b = blockIdx.y;
  const int m = blockIdx.x * 256 + threadIdx.x;
  const int mq = m < Mq ? m : (Mq - 1);
  const float* q = Q + ((size_t)b * Mq + mq) * 3;
  const float qx = q[0], qy = q[1], qz = q[2];
  float bd[KN]; int bi[KN];
#pragma unroll
  for (int i = 0; i < KN; ++i) { bd[i] = 1e30f; bi[i] = 0; }
  __shared__ float sx[256], sy[256], sz[256];
  for (int t0 = 0; t0 < Ns; t0 += 256) {
    const int tn = (Ns - t0) < 256 ? (Ns - t0) : 256;
    __syncthreads();
    if ((int)threadIdx.x < tn) {
      const float* s = S + ((size_t)b * Ns + t0 + threadIdx.x) * 3;
      sx[threadIdx.x] = s[0]; sy[threadIdx.x] = s[1]; sz[threadIdx.x] = s[2];
    }
    if (t0 + 256 < Ns)   // speculative prefetch of next tile (global_prefetch_b8)
      __builtin_prefetch(S + ((size_t)b * Ns + t0 + 256) * 3, 0, 0);
    __syncthreads();
    for (int j = 0; j < tn; ++j) {
      const float dx = qx - sx[j], dy = qy - sy[j], dz = qz - sz[j];
      const float d = dx * dx + dy * dy + dz * dz;
      if (d < bd[KN - 1]) {
        int p = KN - 1;
        while (p > 0 && bd[p - 1] > d) { bd[p] = bd[p - 1]; bi[p] = bi[p - 1]; --p; }
        bd[p] = d; bi[p] = t0 + j;
      }
    }
  }
  if (m < Mq) {
#pragma unroll
    for (int i = 0; i < KN; ++i) out[((size_t)b * Mq + m) * KN + i] = bi[i];
  }
}

// ---------------------------------------------------------------------------
// Farthest point sampling: one block per batch, block-wide argmax each step.
// ---------------------------------------------------------------------------
__global__ __launch_bounds__(256)
void fps_kernel(const float* __restrict__ P, int* __restrict__ fidx,
                float* __restrict__ dmin, int Np, int m) {
  const int b = blockIdx.x;
  const float* pts = P + (size_t)b * Np * 3;
  float* dm = dmin + (size_t)b * Np;
  const int tid = threadIdx.x;
  for (int i = tid; i < Np; i += 256) dm[i] = 1e10f;
  __shared__ float sred[256];
  __shared__ int   sidx[256];
  __shared__ int   s_last;
  if (tid == 0) { fidx[(size_t)b * m] = 0; s_last = 0; }
  __syncthreads();
  for (int it = 1; it < m; ++it) {
    const int last = s_last;
    const float lx = pts[(size_t)last * 3 + 0];
    const float ly = pts[(size_t)last * 3 + 1];
    const float lz = pts[(size_t)last * 3 + 2];
    float bestd = -1.0f; int besti = 0;
    for (int i = tid; i < Np; i += 256) {
      const float dx = pts[(size_t)i * 3 + 0] - lx;
      const float dy = pts[(size_t)i * 3 + 1] - ly;
      const float dz = pts[(size_t)i * 3 + 2] - lz;
      const float d = dx * dx + dy * dy + dz * dz;
      const float v = fminf(dm[i], d);
      dm[i] = v;
      if (v > bestd) { bestd = v; besti = i; }
    }
    sred[tid] = bestd; sidx[tid] = besti;
    __syncthreads();
    for (int s = 128; s > 0; s >>= 1) {
      if (tid < s) {
        if (sred[tid + s] > sred[tid] ||
            (sred[tid + s] == sred[tid] && sidx[tid + s] < sidx[tid])) {
          sred[tid] = sred[tid + s]; sidx[tid] = sidx[tid + s];
        }
      }
      __syncthreads();
    }
    if (tid == 0) { fidx[(size_t)b * m + it] = sidx[0]; s_last = sidx[0]; }
    __syncthreads();
  }
}

// ---------------------------------------------------------------------------
// GroupNorm over [B,N,C] normalizing (N, C/G) per (b, group); optional ReLU.
// ---------------------------------------------------------------------------
template<bool RELU>
__global__ __launch_bounds__(256)
void groupnorm_kernel(const float* __restrict__ X, float* __restrict__ Y,
                      const float* __restrict__ gamma, const float* __restrict__ beta,
                      int N, int C, int G) {
  const int g = blockIdx.x, b = blockIdx.y;
  const int Cg = C / G;
  const size_t base = (size_t)b * N * C;
  const int tid = threadIdx.x;
  const int tot = N * Cg;
  float s = 0.0f, s2 = 0.0f;
  for (int t = tid; t < tot; t += 256) {
    const int n = t / Cg, cc = t % Cg;
    const float v = X[base + (size_t)n * C + g * Cg + cc];
    s += v; s2 += v * v;
  }
  __shared__ float sh[256], sh2[256];
  sh[tid] = s; sh2[tid] = s2;
  __syncthreads();
  for (int st = 128; st > 0; st >>= 1) {
    if (tid < st) { sh[tid] += sh[tid + st]; sh2[tid] += sh2[tid + st]; }
    __syncthreads();
  }
  const float mean = sh[0] / (float)tot;
  const float var  = sh2[0] / (float)tot - mean * mean;
  const float rstd = rsqrtf(var + 1e-5f);
  for (int t = tid; t < tot; t += 256) {
    const int n = t / Cg, cc = t % Cg;
    const int c = g * Cg + cc;
    float v = (X[base + (size_t)n * C + c] - mean) * rstd * gamma[c] + beta[c];
    if (RELU) v = fmaxf(v, 0.0f);
    Y[base + (size_t)n * C + c] = v;
  }
}

// ---------------------------------------------------------------------------
// Elementwise / gather helpers
// ---------------------------------------------------------------------------
__global__ void transpose_bcn_kernel(const float* __restrict__ X, float* __restrict__ Y,
                                     int C, int N, int total) {
  int t = blockIdx.x * blockDim.x + threadIdx.x;
  if (t >= total) return;
  const int c = t % C; int rem = t / C;
  const int n = rem % N; const int b = rem / N;
  Y[t] = X[((size_t)b * C + c) * N + n];
}

__global__ void gather_rows_kernel(const float* __restrict__ X, const int* __restrict__ idx,
                                   float* __restrict__ Y, int Nin, int Nout, int C, int total) {
  int t = blockIdx.x * blockDim.x + threadIdx.x;
  if (t >= total) return;
  const int c = t % C; const int rem = t / C;
  const int i = rem % Nout; const int b = rem / Nout;
  Y[t] = X[((size_t)b * Nin + idx[rem]) * C + c];
}

__global__ void rel_pos_kernel(const float* __restrict__ P, const int* __restrict__ idx,
                               float* __restrict__ rel, int N, int K, int total) {
  int t = blockIdx.x * blockDim.x + threadIdx.x;
  if (t >= total) return;
  const int rem = t / K;
  const int n = rem % N; const int b = rem / N;
  const int s = idx[t];
  const float* pq = P + ((size_t)b * N + n) * 3;
  const float* ps = P + ((size_t)b * N + s) * 3;
  rel[(size_t)t * 3 + 0] = pq[0] - ps[0];
  rel[(size_t)t * 3 + 1] = pq[1] - ps[1];
  rel[(size_t)t * 3 + 2] = pq[2] - ps[2];
}

__global__ void attn_pre_kernel(const float* __restrict__ Q, const float* __restrict__ Kf,
                                const float* __restrict__ PE, const int* __restrict__ idx,
                                float* __restrict__ A0, int N, int K, int C, int total) {
  int t = blockIdx.x * blockDim.x + threadIdx.x;
  if (t >= total) return;
  const int c = t % C; const int rem = t / C;     // rem = (b*N+n)*K + j
  const int bn = rem / K;                         // b*N + n
  const int b = bn / N;
  const int s = idx[rem];
  A0[t] = Q[(size_t)bn * C + c] - Kf[((size_t)b * N + s) * C + c] + PE[t];
}

__global__ void softmax_k_kernel(float* __restrict__ A, int K, int C, int total) {
  int t = blockIdx.x * blockDim.x + threadIdx.x;
  if (t >= total) return;                          // t over (b*N+n)*C + c
  const int c = t % C; const int bn = t / C;
  const size_t base = (size_t)bn * K * C + c;
  float mx = -1e30f;
  for (int j = 0; j < K; ++j) mx = fmaxf(mx, A[base + (size_t)j * C]);
  float s = 0.0f;
  for (int j = 0; j < K; ++j) {
    const float e = expf(A[base + (size_t)j * C] - mx);
    A[base + (size_t)j * C] = e;
    s += e;
  }
  const float inv = 1.0f / s;
  for (int j = 0; j < K; ++j) A[base + (size_t)j * C] *= inv;
}

__global__ void attn_out_kernel(const float* __restrict__ A, const float* __restrict__ V,
                                const float* __restrict__ PE, const int* __restrict__ idx,
                                float* __restrict__ Y, int N, int K, int C, int total) {
  int t = blockIdx.x * blockDim.x + threadIdx.x;
  if (t >= total) return;
  const int c = t % C; const int bn = t / C;
  const int b = bn / N;
  float acc = 0.0f;
  for (int j = 0; j < K; ++j) {
    const int s = idx[(size_t)bn * K + j];
    const size_t ai = ((size_t)bn * K + j) * C + c;
    acc += A[ai] * (V[((size_t)b * N + s) * C + c] + PE[ai]);
  }
  Y[t] = acc;
}

__global__ void add_relu_kernel(const float* __restrict__ X, const float* __restrict__ Yin,
                                float* __restrict__ Y, int total) {
  int t = blockIdx.x * blockDim.x + threadIdx.x;
  if (t >= total) return;
  Y[t] = fmaxf(X[t] + Yin[t], 0.0f);
}

__global__ void td_feat_kernel(const float* __restrict__ P, const float* __restrict__ X,
                               const float* __restrict__ NP, const int* __restrict__ idx,
                               float* __restrict__ feat, int Nin, int Mo, int K, int Cin,
                               int total) {
  int t = blockIdx.x * blockDim.x + threadIdx.x;
  if (t >= total) return;                          // t over (b, i, j)
  const int rem = t / K;
  const int i = rem % Mo; const int b = rem / Mo;
  const int s = idx[t];
  const int Cf = 3 + Cin;
  const float* ps = P + ((size_t)b * Nin + s) * 3;
  const float* np = NP + ((size_t)b * Mo + i) * 3;
  float* f = feat + (size_t)t * Cf;
  f[0] = ps[0] - np[0]; f[1] = ps[1] - np[1]; f[2] = ps[2] - np[2];
  const float* xs = X + ((size_t)b * Nin + s) * Cin;
  for (int c = 0; c < Cin; ++c) f[3 + c] = xs[c];
}

__global__ void max_over_k_kernel(const float* __restrict__ H, float* __restrict__ Y,
                                  int K, int C, int total) {
  int t = blockIdx.x * blockDim.x + threadIdx.x;
  if (t >= total) return;                          // t over (b*M+i)*C + c
  const int c = t % C; const int bi = t / C;
  const size_t base = (size_t)bi * K * C + c;
  float m = -1e30f;
  for (int j = 0; j < K; ++j) m = fmaxf(m, H[base + (size_t)j * C]);
  Y[t] = m;
}

__global__ void interp_kernel(const float* __restrict__ Pf, const float* __restrict__ Pc,
                              const int* __restrict__ idx, const float* __restrict__ Yc,
                              const float* __restrict__ Ysk, float* __restrict__ Out,
                              int Nf, int Nc, int C, int total) {
  int t = blockIdx.x * blockDim.x + threadIdx.x;
  if (t >= total) return;
  const int c = t % C; const int rem = t / C;
  const int n = rem % Nf; const int b = rem / Nf;
  const float* pf = Pf + ((size_t)b * Nf + n) * 3;
  float w[3]; int jj[3]; float tot = 0.0f;
#pragma unroll
  for (int j = 0; j < 3; ++j) {
    jj[j] = idx[((size_t)b * Nf + n) * 3 + j];
    const float* pc = Pc + ((size_t)b * Nc + jj[j]) * 3;
    const float dx = pf[0] - pc[0], dy = pf[1] - pc[1], dz = pf[2] - pc[2];
    const float d = dx * dx + dy * dy + dz * dz;
    w[j] = 1.0f / (d + 1e-8f);
    tot += w[j];
  }
  float acc = Ysk[t];
  const float inv = 1.0f / tot;
#pragma unroll
  for (int j = 0; j < 3; ++j)
    acc += w[j] * inv * Yc[((size_t)b * Nc + jj[j]) * C + c];
  Out[t] = acc;
}

// ---------------------------------------------------------------------------
// Host orchestration
// ---------------------------------------------------------------------------
struct BlockP {
  const float *lin1, *g1, *b1, *Wq, *Wk, *Wv, *pW1, *pb1, *pW2, *pb2;
  const float *aW1, *ab1, *aW2, *ab2, *g2, *b2, *lin3, *g3, *b3;
};

extern "C" void kernel_launch(void* const* d_in, const int* in_sizes, int n_in,
                              void* d_out, int out_size, void* d_ws, size_t ws_size,
                              hipStream_t stream) {
  (void)in_sizes; (void)n_in; (void)out_size; (void)ws_size;

  const int B = 2;
  const int N1 = 8192, N2 = 2048, N3 = 512, N4 = 128, N5 = 32;

  const float* p1   = (const float*)d_in[0];
  const float* x1in = (const float*)d_in[1];
  float* out = (float*)d_out;

  // --- flattened parameter traversal (setup_inputs dict order) ---
  int pi = 2;
  auto F = [&]() { return (const float*)d_in[pi++]; };
  auto loadBlock = [&]() {
    BlockP p;
    p.lin1 = F(); p.g1 = F(); p.b1 = F();
    p.Wq = F(); p.Wk = F(); p.Wv = F();
    p.pW1 = F(); p.pb1 = F(); p.pW2 = F(); p.pb2 = F();
    p.aW1 = F(); p.ab1 = F(); p.aW2 = F(); p.ab2 = F();
    p.g2 = F(); p.b2 = F();
    p.lin3 = F(); p.g3 = F(); p.b3 = F();
    return p;
  };
  const float* in_W = F(); const float* in_g = F(); const float* in_b = F();
  BlockP enc1 = loadBlock();
  const float* d12_W = F(); const float* d12_b = F();
  BlockP enc2 = loadBlock();
  const float* d23_W = F(); const float* d23_b = F();
  BlockP enc3 = loadBlock();
  const float* d34_W = F(); const float* d34_b = F();
  BlockP enc4 = loadBlock();
  const float* d45_W = F(); const float* d45_b = F();
  BlockP enc5 = loadBlock();
  const float* agg_W = F(); const float* agg_b = F();
  const float* dm_W1 = F(); const float* dm_g1 = F(); const float* dm_b1 = F();
  const float* dm_W2 = F(); const float* dm_g2 = F(); const float* dm_b2 = F();
  BlockP dec5 = loadBlock();
  const float* u54_Wi = F(); const float* u54_bi = F();
  const float* u54_Ws = F(); const float* u54_bs = F();
  BlockP dec4 = loadBlock();
  const float* u43_Wi = F(); const float* u43_bi = F();
  const float* u43_Ws = F(); const float* u43_bs = F();
  BlockP dec3 = loadBlock();
  const float* u32_Wi = F(); const float* u32_bi = F();
  const float* u32_Ws = F(); const float* u32_bs = F();
  BlockP dec2 = loadBlock();
  const float* om_W1 = F(); const float* om_g = F(); const float* om_b = F();
  const float* om_W2 = F(); const float* om_b2 = F();

  // --- workspace bump allocator ---
  size_t ws_off = 0;
  auto alloc = [&](size_t n) -> float* {
    ws_off = (ws_off + 255) & ~(size_t)255;
    float* p = (float*)((char*)d_ws + ws_off);
    ws_off += n * sizeof(float);
    return p;
  };
  auto alloci = [&](size_t n) -> int* {
    ws_off = (ws_off + 255) & ~(size_t)255;
    int* p = (int*)((char*)d_ws + ws_off);
    ws_off += n * sizeof(int);
    return p;
  };

  // persistent cross-stage tensors
  float* xt  = alloc((size_t)B * N1 * 6);
  float* x0  = alloc((size_t)B * N1 * 16);
  float* x1e = alloc((size_t)B * N1 * 16);
  float* p2  = alloc((size_t)B * N2 * 3);
  float* x2d = alloc((size_t)B * N2 * 16);
  float* x2  = alloc((size_t)B * N2 * 16);
  float* p3  = alloc((size_t)B * N3 * 3);
  float* x3d = alloc((size_t)B * N3 * 32);
  float* x3  = alloc((size_t)B * N3 * 32);
  float* p4  = alloc((size_t)B * N4 * 3);
  float* x4d = alloc((size_t)B * N4 * 64);
  float* x4  = alloc((size_t)B * N4 * 64);
  float* p5  = alloc((size_t)B * N5 * 3);
  float* x5d = alloc((size_t)B * N5 * 64);
  float* x5  = alloc((size_t)B * N5 * 64);
  float* d5a = alloc((size_t)B * N5 * 64);
  float* d5b = alloc((size_t)B * N5 * 64);
  float* y5  = alloc((size_t)B * N5 * 64);
  float* y4u = alloc((size_t)B * N4 * 64);
  float* y4  = alloc((size_t)B * N4 * 64);
  float* y3u = alloc((size_t)B * N3 * 32);
  float* y3  = alloc((size_t)B * N3 * 32);
  float* y2u = alloc((size_t)B * N2 * 16);
  float* y2d = alloc((size_t)B * N2 * 16);
  float* t1  = alloc((size_t)B * N2 * 16);
  float* t2  = alloc((size_t)B * N2 * 16);
  const size_t ws_base = ws_off;
  auto reset = [&]() { ws_off = ws_base; };

  // --- launch helpers ---
  auto gemm = [&](const float* X, const float* W, const float* bias, float* Y,
                  int M, int K, int N, bool relu) {
    dim3 grid(cdiv(M, 64), cdiv(N, 16));
    if (N == 16) {
      if (bias) { if (relu) gemm_wmma_kernel<true,  true,  16><<<grid, 128, 0, stream>>>(X, W, bias, Y, M, K);
                  else      gemm_wmma_kernel<false, true,  16><<<grid, 128, 0, stream>>>(X, W, bias, Y, M, K); }
      else      { if (relu) gemm_wmma_kernel<true,  false, 16><<<grid, 128, 0, stream>>>(X, W, nullptr, Y, M, K);
                  else      gemm_wmma_kernel<false, false, 16><<<grid, 128, 0, stream>>>(X, W, nullptr, Y, M, K); }
    } else if (N == 32) {
      if (bias) { if (relu) gemm_wmma_kernel<true,  true,  32><<<grid, 128, 0, stream>>>(X, W, bias, Y, M, K);
                  else      gemm_wmma_kernel<false, true,  32><<<grid, 128, 0, stream>>>(X, W, bias, Y, M, K); }
      else      { if (relu) gemm_wmma_kernel<true,  false, 32><<<grid, 128, 0, stream>>>(X, W, nullptr, Y, M, K);
                  else      gemm_wmma_kernel<false, false, 32><<<grid, 128, 0, stream>>>(X, W, nullptr, Y, M, K); }
    } else {
      if (bias) { if (relu) gemm_wmma_kernel<true,  true,  64><<<grid, 128, 0, stream>>>(X, W, bias, Y, M, K);
                  else      gemm_wmma_kernel<false, true,  64><<<grid, 128, 0, stream>>>(X, W, bias, Y, M, K); }
      else      { if (relu) gemm_wmma_kernel<true,  false, 64><<<grid, 128, 0, stream>>>(X, W, nullptr, Y, M, K);
                  else      gemm_wmma_kernel<false, false, 64><<<grid, 128, 0, stream>>>(X, W, nullptr, Y, M, K); }
    }
  };
  auto gn = [&](const float* X, float* Y, const float* g, const float* b,
                int N, int C, int G, bool relu) {
    dim3 grid(G, B);
    if (relu) groupnorm_kernel<true ><<<grid, 256, 0, stream>>>(X, Y, g, b, N, C, G);
    else      groupnorm_kernel<false><<<grid, 256, 0, stream>>>(X, Y, g, b, N, C, G);
  };
  auto knn = [&](const float* q, const float* s, int* idx, int Mq, int Ns, int K) {
    dim3 grid(cdiv(Mq, 256), B);
    if (K == 3)      knn_kernel<3 ><<<grid, 256, 0, stream>>>(q, s, idx, Mq, Ns);
    else if (K == 8) knn_kernel<8 ><<<grid, 256, 0, stream>>>(q, s, idx, Mq, Ns);
    else             knn_kernel<16><<<grid, 256, 0, stream>>>(q, s, idx, Mq, Ns);
  };

  auto pt_attn = [&](const BlockP& pr, const float* p, const float* x, float* outb,
                     int N, int C, int K) {
    const int BN = B * N;
    const int BNK = BN * K;
    float* q   = alloc((size_t)BN * C);
    float* kf  = alloc((size_t)BN * C);
    float* v   = alloc((size_t)BN * C);
    int*   idx = alloci((size_t)BNK);
    float* rel = alloc((size_t)BNK * 3);
    float* pe  = alloc((size_t)BNK * C);
    float* a0  = alloc((size_t)BNK * C);
    float* a1  = alloc((size_t)BNK * C);
    gemm(x, pr.Wq, nullptr, q,  BN, C, C, false);
    gemm(x, pr.Wk, nullptr, kf, BN, C, C, false);
    gemm(x, pr.Wv, nullptr, v,  BN, C, C, false);
    knn(p, p, idx, N, N, K);
    { int tot = BNK;
      rel_pos_kernel<<<cdiv(tot, 256), 256, 0, stream>>>(p, idx, rel, N, K, tot); }
    gemm(rel, pr.pW1, pr.pb1, a0, BNK, 3, C, true);
    gemm(a0,  pr.pW2, pr.pb2, pe, BNK, C, C, false);
    { int tot = BNK * C;
      attn_pre_kernel<<<cdiv(tot, 256), 256, 0, stream>>>(q, kf, pe, idx, a0, N, K, C, tot); }
    gemm(a0, pr.aW1, pr.ab1, a1, BNK, C, C, true);
    gemm(a1, pr.aW2, pr.ab2, a0, BNK, C, C, false);
    { int tot = BN * C;
      softmax_k_kernel<<<cdiv(tot, 256), 256, 0, stream>>>(a0, K, C, tot); }
    { int tot = BN * C;
      attn_out_kernel<<<cdiv(tot, 256), 256, 0, stream>>>(a0, v, pe, idx, outb, N, K, C, tot); }
  };

  auto pt_block = [&](const BlockP& pr, const float* p, const float* xin, float* xout,
                      int N, int C, int K, int G) {
    const int BN = B * N;
    float* ya = alloc((size_t)BN * C);
    float* yb = alloc((size_t)BN * C);
    gemm(xin, pr.lin1, nullptr, ya, BN, C, C, false);
    gn(ya, yb, pr.g1, pr.b1, N, C, G, true);
    pt_attn(pr, p, yb, ya, N, C, K);
    gn(ya, yb, pr.g2, pr.b2, N, C, G, true);
    gemm(yb, pr.lin3, nullptr, ya, BN, C, C, false);
    gn(ya, yb, pr.g3, pr.b3, N, C, G, false);
    { int tot = BN * C;
      add_relu_kernel<<<cdiv(tot, 256), 256, 0, stream>>>(xin, yb, xout, tot); }
  };

  auto t_down = [&](const float* W, const float* bias, const float* p, const float* x,
                    float* newp, float* newx, int Nin, int Cin, int Cout, int K) {
    const int Mo = Nin / 4;
    int*   fidx = alloci((size_t)B * Mo);
    float* dmin = alloc((size_t)B * Nin);
    fps_kernel<<<B, 256, 0, stream>>>(p, fidx, dmin, Nin, Mo);
    { int tot = B * Mo * 3;
      gather_rows_kernel<<<cdiv(tot, 256), 256, 0, stream>>>(p, fidx, newp, Nin, Mo, 3, tot); }
    int* idx = alloci((size_t)B * Mo * K);
    knn(newp, p, idx, Mo, Nin, K);
    const int Cf = 3 + Cin;
    float* feat = alloc((size_t)B * Mo * K * Cf);
    { int tot = B * Mo * K;
      td_feat_kernel<<<cdiv(tot, 256), 256, 0, stream>>>(p, x, newp, idx, feat, Nin, Mo, K, Cin, tot); }
    float* h = alloc((size_t)B * Mo * K * Cout);
    gemm(feat, W, bias, h, B * Mo * K, Cf, Cout, true);
    { int tot = B * Mo * Cout;
      max_over_k_kernel<<<cdiv(tot, 256), 256, 0, stream>>>(h, newx, K, Cout, tot); }
  };

  auto t_up = [&](const float* Wi, const float* bi, const float* Ws, const float* bs,
                  const float* pc, const float* xc, const float* pf, const float* xskip,
                  float* o, int Nc, int Nf, int Cc, int Cf) {
    float* ysk = alloc((size_t)B * Nf * Cf);
    float* yc  = alloc((size_t)B * Nc * Cf);
    gemm(xskip, Ws, bs, ysk, B * Nf, Cf, Cf, false);
    gemm(xc,    Wi, bi, yc,  B * Nc, Cc, Cf, false);
    int* idx = alloci((size_t)B * Nf * 3);
    knn(pf, pc, idx, Nf, Nc, 3);
    { int tot = B * Nf * Cf;
      interp_kernel<<<cdiv(tot, 256), 256, 0, stream>>>(pf, pc, idx, yc, ysk, o, Nf, Nc, Cf, tot); }
  };

  // ================== forward pass ==================
  // in_mlp: transpose [B,6,N]->[B,N,6], linear 6->16, GN(1)+ReLU
  reset();
  { int tot = B * N1 * 6;
    transpose_bcn_kernel<<<cdiv(tot, 256), 256, 0, stream>>>(x1in, xt, 6, N1, tot); }
  { float* tmp = alloc((size_t)B * N1 * 16);
    gemm(xt, in_W, nullptr, tmp, B * N1, 6, 16, false);
    gn(tmp, x0, in_g, in_b, N1, 16, 1, true); }

  reset(); pt_block(enc1, p1, x0, x1e, N1, 16, 8, 1);
  reset(); t_down(d12_W, d12_b, p1, x1e, p2, x2d, N1, 16, 16, 16);
  reset(); pt_block(enc2, p2, x2d, x2, N2, 16, 16, 1);
  reset(); t_down(d23_W, d23_b, p2, x2, p3, x3d, N2, 16, 32, 16);
  reset(); pt_block(enc3, p3, x3d, x3, N3, 32, 16, 2);
  reset(); t_down(d34_W, d34_b, p3, x3, p4, x4d, N3, 32, 64, 16);
  reset(); pt_block(enc4, p4, x4d, x4, N4, 64, 16, 4);
  reset(); t_down(d45_W, d45_b, p4, x4, p5, x5d, N4, 64, 64, 16);
  reset(); pt_block(enc5, p5, x5d, x5, N5, 64, 16, 4);

  // scene = max_n relu(x5 @ aggW + aggb)   -> out + 65536
  reset();
  { float* h = alloc((size_t)B * N5 * 64);
    gemm(x5, agg_W, agg_b, h, B * N5, 64, 64, true);
    int tot = B * 64;
    max_over_k_kernel<<<cdiv(tot, 256), 256, 0, stream>>>(h, out + (size_t)B * N2 * 16, N5, 64, tot); }

  // dec_mlp
  reset();
  { float* tmp = alloc((size_t)B * N5 * 64);
    gemm(x5, dm_W1, nullptr, tmp, B * N5, 64, 64, false);
    gn(tmp, d5a, dm_g1, dm_b1, N5, 64, 4, true);
    gemm(d5a, dm_W2, nullptr, tmp, B * N5, 64, 64, false);
    gn(tmp, d5b, dm_g2, dm_b2, N5, 64, 4, true); }

  reset(); pt_block(dec5, p5, d5b, y5, N5, 64, 16, 4);
  reset(); t_up(u54_Wi, u54_bi, u54_Ws, u54_bs, p5, y5, p4, x4, y4u, N5, N4, 64, 64);
  reset(); pt_block(dec4, p4, y4u, y4, N4, 64, 16, 4);
  reset(); t_up(u43_Wi, u43_bi, u43_Ws, u43_bs, p4, y4, p3, x3, y3u, N4, N3, 64, 32);
  reset(); pt_block(dec3, p3, y3u, y3, N3, 32, 16, 2);
  reset(); t_up(u32_Wi, u32_bi, u32_Ws, u32_bs, p3, y3, p2, x2, y2u, N3, N2, 32, 16);
  reset(); pt_block(dec2, p2, y2u, y2d, N2, 16, 16, 1);

  // out_mlp -> y into out[0 .. B*N2*16)
  reset();
  gemm(y2d, om_W1, nullptr, t1, B * N2, 16, 16, false);
  gn(t1, t2, om_g, om_b, N2, 16, 1, true);
  gemm(t2, om_W2, om_b2, out, B * N2, 16, 16, false);

  // p2 -> out tail (after y and scene)
  hipMemcpyAsync(out + (size_t)B * N2 * 16 + (size_t)B * 64, p2,
                 sizeof(float) * (size_t)B * N2 * 3, hipMemcpyDeviceToDevice, stream);
}
